// TransformerDualAttentionLayer_40621800686145
// MI455X (gfx1250) — compile-verified, matrix operand order checked by reading, not run on hardware
//
#include <hip/hip_runtime.h>
#include <hip/hip_bf16.h>
#include <math.h>

typedef __attribute__((ext_vector_type(16))) __bf16 v16bf;
typedef __attribute__((ext_vector_type(8)))  float  v8f;
typedef __attribute__((__vector_size__(16))) int v4i32;
typedef unsigned short u16;
typedef unsigned char  u8;

static constexpr long long NB  = 16;
static constexpr long long NLX = 1024;
static constexpr long long NLY = 2048;
static constexpr long long ND  = 1024;
static constexpr long long NH  = 8;
static constexpr long long NDH = 128;
static constexpr float ATT_SCALE = 0.08838834764831845f; // 1/sqrt(128)

#define DEVI __device__ __forceinline__

// gfx1250 async global->LDS path (probe-guarded; falls back to sync copy)
#if defined(__AMDGCN__) && __has_builtin(__builtin_amdgcn_global_load_async_to_lds_b128) && __has_builtin(__builtin_amdgcn_s_wait_asynccnt)
#define USE_ASYNC_LDS 1
#else
#define USE_ASYNC_LDS 0
#endif

DEVI void cp16(const u16* g, u16* l) {
#if USE_ASYNC_LDS
  __builtin_amdgcn_global_load_async_to_lds_b128(
      (v4i32*)const_cast<u16*>(g), (v4i32*)l, 0, 0);
#else
  *(uint4*)l = *(const uint4*)g;
#endif
}

DEVI void cp_wait() {
#if USE_ASYNC_LDS
  __builtin_amdgcn_s_wait_asynccnt(0);
#endif
}

DEVI u16 f2bf(float f) {  // round-to-nearest-even f32 -> bf16
  unsigned int u = __float_as_uint(f);
  unsigned int r = u + 0x7FFFu + ((u >> 16) & 1u);
  return (u16)(r >> 16);
}

// ---------------------------------------------------------------------------
// Tiled bf16 GEMM:  C[z] = alpha * A[z] x B[z]  (+bias)(+residual)(+relu)
//   BT=true : B supplied as [N,K] row-major (NT gemm, e.g. Q @ K^T)
//   BT=false: B supplied as [K,N] row-major (NN gemm), transposed into LDS
//   EPI: 0=none 1=+bias 2=+bias+residual 3=relu(+bias)
//   OUTB: store bf16 (else f32)
// Block tile 128x128x32, 4 waves, 64x64 per wave, v_wmma_f32_16x16x32_bf16.
// ---------------------------------------------------------------------------
static constexpr int TBM = 128, TBN = 128, TBK = 32, LDSS = TBK + 8;

union FragBF { v16bf v; uint4 q[2]; };

template<bool BT, int EPI, bool OUTB>
__global__ __launch_bounds__(128)
void gemm_bf16_k(const u16* __restrict__ A, long long lda, long long saz,
                 const u16* __restrict__ Bm, long long ldb, long long sbz,
                 void* __restrict__ Cv, long long ldc, long long scz,
                 const float* __restrict__ bias,
                 const float* __restrict__ Rf, long long ldr, long long srz,
                 int K, float alpha)
{
  __shared__ __align__(16) u16 As[2][TBM * LDSS];
  __shared__ __align__(16) u16 Bs[2][TBN * LDSS];

  const int tid  = threadIdx.x;
  const int lane = tid & 31;
  const int wave = tid >> 5;
  const int wm = (wave >> 1) * 64;
  const int wn = (wave & 1) * 64;
  const long long bm0 = (long long)blockIdx.x * TBM;
  const long long bn0 = (long long)blockIdx.y * TBN;
  const long long z   = blockIdx.z;

  A  += z * saz;
  Bm += z * sbz;

  v8f acc[4][4];
#pragma unroll
  for (int i = 0; i < 4; ++i)
#pragma unroll
    for (int j = 0; j < 4; ++j)
#pragma unroll
      for (int e = 0; e < 8; ++e) acc[i][j][e] = 0.0f;

  const int ktiles = K / TBK;

  auto load_tiles = [&](int kt, int buf) {
    {
      const u16* g = A + (bm0 + tid) * lda + (long long)kt * TBK;
      u16* l = &As[buf][tid * LDSS];
      cp16(g, l); cp16(g + 8, l + 8); cp16(g + 16, l + 16); cp16(g + 24, l + 24);
    }
    if (BT) {
      const u16* g = Bm + (bn0 + tid) * ldb + (long long)kt * TBK;
      u16* l = &Bs[buf][tid * LDSS];
      cp16(g, l); cp16(g + 8, l + 8); cp16(g + 16, l + 16); cp16(g + 24, l + 24);
    } else {
#pragma unroll
      for (int i = 0; i < 4; ++i) {          // transpose [K,N] tile into Bs[n][k]
        int c  = tid + i * 128;
        int k  = c >> 4;
        int ns = (c & 15) * 8;
        union { uint4 v; u16 s[8]; } un;
        un.v = *(const uint4*)(Bm + ((long long)kt * TBK + k) * ldb + bn0 + ns);
#pragma unroll
        for (int j = 0; j < 8; ++j) Bs[buf][(ns + j) * LDSS + k] = un.s[j];
      }
    }
  };

  load_tiles(0, 0);
  cp_wait();
  __syncthreads();

  for (int kt = 0; kt < ktiles; ++kt) {
    const int cur = kt & 1;
    if (kt + 1 < ktiles) load_tiles(kt + 1, cur ^ 1);
    if (kt + 2 < ktiles)
      __builtin_prefetch(A + (bm0 + tid) * lda + (long long)(kt + 2) * TBK, 0, 1);

    FragBF fa[4], fb[4];
    const int r16 = lane & 15;
    const int kb  = (lane >> 4) * 8;   // K-half select per WMMA 16-bit layout
#pragma unroll
    for (int r = 0; r < 4; ++r) {
      const u16* base = &As[cur][(wm + r * 16 + r16) * LDSS];
      fa[r].q[0] = *(const uint4*)(base + kb);
      fa[r].q[1] = *(const uint4*)(base + kb + 16);
    }
#pragma unroll
    for (int c = 0; c < 4; ++c) {
      const u16* base = &Bs[cur][(wn + c * 16 + r16) * LDSS];
      fb[c].q[0] = *(const uint4*)(base + kb);
      fb[c].q[1] = *(const uint4*)(base + kb + 16);
    }
#pragma unroll
    for (int r = 0; r < 4; ++r)
#pragma unroll
      for (int c = 0; c < 4; ++c)
        acc[r][c] = __builtin_amdgcn_wmma_f32_16x16x32_bf16(
            false, fa[r].v, false, fb[c].v, (short)0, acc[r][c], false, false);

    cp_wait();
    __syncthreads();
  }

  // epilogue (C/D layout: lanes 0-15 rows e, lanes 16-31 rows 8+e)
  const long long czo = z * scz;
  const long long rzo = z * srz;
  const int coll = lane & 15;
  const int rofs = (lane >> 4) * 8;
#pragma unroll
  for (int r = 0; r < 4; ++r) {
#pragma unroll
    for (int c = 0; c < 4; ++c) {
      const long long gc = bn0 + wn + c * 16 + coll;
      const float bv = (EPI >= 1) ? bias[gc] : 0.0f;
#pragma unroll
      for (int e = 0; e < 8; ++e) {
        const long long gr = bm0 + wm + r * 16 + rofs + e;
        float v = acc[r][c][e] * alpha;
        if (EPI >= 1) v += bv;
        if (EPI == 2) v += Rf[rzo + gr * ldr + gc];
        if (EPI == 3) v = fmaxf(v, 0.0f);
        if (OUTB) ((u16*)Cv)[czo + gr * ldc + gc] = f2bf(v);
        else      ((float*)Cv)[czo + gr * ldc + gc] = v;
      }
    }
  }
}

// ---------------------------------------------------------------------------
// elementwise / reduction kernels
// ---------------------------------------------------------------------------
DEVI float block_reduce(float v, bool do_max) {
#pragma unroll
  for (int o = 16; o > 0; o >>= 1) {
    float other = __shfl_xor(v, o, 32);
    v = do_max ? fmaxf(v, other) : (v + other);
  }
  __shared__ float tmp[8];
  __shared__ float res;
  const int w = threadIdx.x >> 5, l = threadIdx.x & 31;
  __syncthreads();
  if (l == 0) tmp[w] = v;
  __syncthreads();
  if (threadIdx.x == 0) {
    float r = tmp[0];
    const int nw = blockDim.x >> 5;
    for (int i = 1; i < nw; ++i) r = do_max ? fmaxf(r, tmp[i]) : (r + tmp[i]);
    res = r;
  }
  __syncthreads();
  return res;
}

__global__ __launch_bounds__(256)
void softmax_rows(const float* __restrict__ S, u16* __restrict__ P,
                  const u8* __restrict__ mask, int C, int rows_per_seg, int mask_stride)
{
  const long long row = blockIdx.x;
  const float* Sr = S + row * C;
  u16* Pr = P + row * C;
  const u8* mr = mask + (long long)(row / rows_per_seg) * mask_stride;
  float loc[8];
  float mx = -INFINITY;
  int n = 0;
  for (int j = threadIdx.x; j < C; j += 256) { float v = mr[j] ? Sr[j] : -INFINITY; loc[n++] = v; mx = fmaxf(mx, v); }
  mx = block_reduce(mx, true);
  float s = 0.0f; n = 0;
  for (int j = threadIdx.x; j < C; j += 256) { float e = __expf(loc[n] - mx); loc[n++] = e; s += e; }
  s = block_reduce(s, false);
  const float inv = 1.0f / s;
  n = 0;
  for (int j = threadIdx.x; j < C; j += 256) Pr[j] = f2bf(loc[n++] * inv);
}

// column softmax of score[b,x,y] over x, output transposed att_yT[b,y,x] (bf16)
__global__ __launch_bounds__(64)
void softmax_cols_T(const float* __restrict__ S, u16* __restrict__ PT,
                    const u8* __restrict__ xmask, int LXc, int LYc)
{
  const long long b  = blockIdx.y;
  const long long yc = (long long)blockIdx.x * 64 + threadIdx.x;
  const float* Sb = S + b * (long long)LXc * LYc;
  const u8* mb = xmask + b * LXc;
  u16* Pb = PT + b * (long long)LYc * LXc + yc * LXc;
  float mx = -INFINITY, sum = 0.0f;
  for (int xi = 0; xi < LXc; ++xi) {
    if (!mb[xi]) continue;
    const float v  = Sb[(long long)xi * LYc + yc];
    const float nm = fmaxf(mx, v);
    sum = sum * __expf(mx - nm) + __expf(v - nm);
    mx = nm;
  }
  const float inv = 1.0f / sum;
  for (int xi = 0; xi < LXc; ++xi) {
    float p = 0.0f;
    if (mb[xi]) p = __expf(Sb[(long long)xi * LYc + yc] - mx) * inv;
    Pb[xi] = f2bf(p);
  }
}

__global__ __launch_bounds__(256)
void layernorm_rows(const float* __restrict__ X, const float* __restrict__ g,
                    const float* __restrict__ bta, float* __restrict__ outF,
                    u16* __restrict__ outB, int Dc)
{
  const long long row = blockIdx.x;
  const float* Xr = X + row * Dc;
  float loc[4];
  float s = 0.0f; int n = 0;
  for (int j = threadIdx.x; j < Dc; j += 256) { loc[n] = Xr[j]; s += loc[n]; ++n; }
  s = block_reduce(s, false);
  const float mu = s / Dc;
  float vs = 0.0f; n = 0;
  for (int j = threadIdx.x; j < Dc; j += 256) { float d = loc[n++] - mu; vs += d * d; }
  vs = block_reduce(vs, false);
  const float rstd = rsqrtf(vs / Dc + 1e-5f);
  n = 0;
  for (int j = threadIdx.x; j < Dc; j += 256) {
    const float o = (loc[n++] - mu) * rstd * g[j] + bta[j];
    outF[row * Dc + j] = o;
    if (outB) outB[row * Dc + j] = f2bf(o);
  }
}

__global__ __launch_bounds__(256)
void cvt_f32_bf16(const float* __restrict__ s, u16* __restrict__ d, long long n)
{
  long long i = (long long)blockIdx.x * 256 + threadIdx.x;
  const long long st = (long long)gridDim.x * 256;
  for (; i < n; i += st) d[i] = f2bf(s[i]);
}

// ---------------------------------------------------------------------------
// host side
// ---------------------------------------------------------------------------
template<bool BT, int EPI, bool OUTB>
static void run_gemm(hipStream_t st,
                     const u16* A, long long lda, long long saz,
                     const u16* Bm, long long ldb, long long sbz,
                     void* C, long long ldc, long long scz,
                     const float* bias,
                     const float* R, long long ldr, long long srz,
                     long long M, long long N, int K, int Z, float alpha)
{
  dim3 grid((unsigned)(M / TBM), (unsigned)(N / TBN), (unsigned)Z);
  gemm_bf16_k<BT, EPI, OUTB><<<grid, dim3(128), 0, st>>>(
      A, lda, saz, Bm, ldb, sbz, C, ldc, scz, bias, R, ldr, srz, K, alpha);
}

extern "C" void kernel_launch(void* const* d_in, const int* in_sizes, int n_in,
                              void* d_out, int out_size, void* d_ws, size_t ws_size,
                              hipStream_t stream)
{
  (void)in_sizes; (void)n_in; (void)out_size; (void)ws_size;

  const float* x = (const float*)d_in[0];
  const float* y = (const float*)d_in[1];
  const u8* xm = (const u8*)d_in[2];
  const u8* ym = (const u8*)d_in[3];
  const float* xp_w = (const float*)d_in[4];
  const float* xp_b = (const float*)d_in[5];
  const float* yp_w = (const float*)d_in[6];
  const float* yp_b = (const float*)d_in[7];
  const float* xn_g = (const float*)d_in[8];
  const float* xn_b = (const float*)d_in[9];
  const float* yn_g = (const float*)d_in[10];
  const float* yn_b = (const float*)d_in[11];
  const float* P[2][12];
  for (int p = 0; p < 2; ++p)
    for (int i = 0; i < 12; ++i) P[p][i] = (const float*)d_in[12 + p * 12 + i];
  // P[p]: 0 qkv_w 1 qkv_b 2 o_w 3 o_b 4 ln1_g 5 ln1_b 6 f1_w 7 f1_b 8 f2_w 9 f2_b 10 ln2_g 11 ln2_b

  char* wp = (char*)d_ws;
  auto alloc = [&](size_t bytes) -> void* {
    void* r = (void*)wp;
    wp += (bytes + 255) & ~(size_t)255;
    return r;
  };

  u16* xb  = (u16*)alloc((size_t)(NB * NLX * ND * 2));
  u16* yb  = (u16*)alloc((size_t)(NB * NLY * ND * 2));
  u16* wxp = (u16*)alloc((size_t)(2 * ND * ND * 2));
  u16* wyp = (u16*)alloc((size_t)(2 * ND * ND * 2));
  u16 *wqkv[2], *wo[2], *wf1[2], *wf2[2];
  for (int p = 0; p < 2; ++p) {
    wqkv[p] = (u16*)alloc((size_t)(ND * 3 * ND * 2));
    wo[p]   = (u16*)alloc((size_t)(ND * ND * 2));
    wf1[p]  = (u16*)alloc((size_t)(ND * ND * 2));
    wf2[p]  = (u16*)alloc((size_t)(ND * ND * 2));
  }
  float* score = (float*)alloc((size_t)(NB * NLX * NLY * 4)); // reused: enc scores (H*LY*LY*4)
  u16* attx  = (u16*)alloc((size_t)(NB * NLX * NLY * 2));     // reused: enc softmax probs
  u16* attyT = (u16*)alloc((size_t)(NB * NLY * NLX * 2));     // reused: enc attention out
  u16* nx2b  = (u16*)alloc((size_t)(NB * NLX * 2 * ND * 2));
  u16* ny2b  = (u16*)alloc((size_t)(NB * NLY * 2 * ND * 2));
  float* pre = (float*)alloc((size_t)(NB * NLY * ND * 4));
  float* xo1 = (float*)alloc((size_t)(NB * NLX * ND * 4));
  u16* xo1b  = (u16*)alloc((size_t)(NB * NLX * ND * 2));
  float* yo1 = (float*)alloc((size_t)(NB * NLY * ND * 4));
  u16* yo1b  = (u16*)alloc((size_t)(NB * NLY * ND * 2));
  u16* qkvb  = (u16*)alloc((size_t)(NB * NLY * 3 * ND * 2));
  float* lnf = (float*)alloc((size_t)(NB * NLY * ND * 4));
  u16* lnb   = (u16*)alloc((size_t)(NB * NLY * ND * 2));
  u16* fbuf  = (u16*)alloc((size_t)(NB * NLY * ND * 2));

  auto cvt = [&](const float* s, u16* d, long long n) {
    cvt_f32_bf16<<<dim3(2048), dim3(256), 0, stream>>>(s, d, n);
  };
  cvt(x, xb, NB * NLX * ND);
  cvt(y, yb, NB * NLY * ND);
  cvt(xp_w, wxp, 2 * ND * ND);
  cvt(yp_w, wyp, 2 * ND * ND);
  for (int p = 0; p < 2; ++p) {
    cvt(P[p][0], wqkv[p], ND * 3 * ND);
    cvt(P[p][2], wo[p], ND * ND);
    cvt(P[p][6], wf1[p], ND * ND);
    cvt(P[p][8], wf2[p], ND * ND);
  }

  // ---- stage 1: shared score + dual softmax ----
  run_gemm<true, 0, false>(stream, xb, ND, NLX * ND, yb, ND, NLY * ND,
                           score, NLY, NLX * NLY, nullptr, nullptr, 0, 0,
                           NLX, NLY, (int)ND, (int)NB, 1.0f);
  softmax_rows<<<dim3((unsigned)(NB * NLX)), dim3(256), 0, stream>>>(
      score, attx, ym, (int)NLY, (int)NLX, (int)NLY);
  softmax_cols_T<<<dim3((unsigned)(NLY / 64), (unsigned)NB), dim3(64), 0, stream>>>(
      score, attyT, xm, (int)NLX, (int)NLY);

  // new_y = att_y^T @ x           -> ny2b[:, :D]
  run_gemm<false, 0, true>(stream, attyT, NLX, NLY * NLX, xb, ND, NLX * ND,
                           ny2b, 2 * ND, NLY * 2 * ND, nullptr, nullptr, 0, 0,
                           NLY, ND, (int)NLX, (int)NB, 1.0f);
  // new_x = att_x @ [y | new_y]   -> nx2b
  run_gemm<false, 0, true>(stream, attx, NLY, NLX * NLY, yb, ND, NLY * ND,
                           nx2b, 2 * ND, NLX * 2 * ND, nullptr, nullptr, 0, 0,
                           NLX, ND, (int)NLY, (int)NB, 1.0f);
  run_gemm<false, 0, true>(stream, attx, NLY, NLX * NLY, ny2b, 2 * ND, NLY * 2 * ND,
                           nx2b + ND, 2 * ND, NLX * 2 * ND, nullptr, nullptr, 0, 0,
                           NLX, ND, (int)NLY, (int)NB, 1.0f);
  // new_y_ = att_y^T @ [x | new_x_] ; second half uses new_x_ = nx2b[:, :D]
  run_gemm<false, 0, true>(stream, attyT, NLX, NLY * NLX, nx2b, 2 * ND, NLX * 2 * ND,
                           ny2b + ND, 2 * ND, NLY * 2 * ND, nullptr, nullptr, 0, 0,
                           NLY, ND, (int)NLX, (int)NB, 1.0f);

  // xo = LN(x + new_x @ xp_w + xp_b) ; yo = LN(y + new_y_ @ yp_w + yp_b)
  run_gemm<false, 2, false>(stream, nx2b, 2 * ND, 0, wxp, ND, 0, pre, ND, 0,
                            xp_b, x, ND, 0, NB * NLX, ND, (int)(2 * ND), 1, 1.0f);
  layernorm_rows<<<dim3((unsigned)(NB * NLX)), dim3(256), 0, stream>>>(pre, xn_g, xn_b, xo1, xo1b, (int)ND);
  run_gemm<false, 2, false>(stream, ny2b, 2 * ND, 0, wyp, ND, 0, pre, ND, 0,
                            yp_b, y, ND, 0, NB * NLY, ND, (int)(2 * ND), 1, 1.0f);
  layernorm_rows<<<dim3((unsigned)(NB * NLY)), dim3(256), 0, stream>>>(pre, yn_g, yn_b, yo1, yo1b, (int)ND);

  // ---- transformer encoder layers ----
  u16* pbuf = attx;
  u16* obuf = attyT;
  auto run_encoder = [&](const float* inF, const u16* inB, const u8* mask, long long L,
                         const float* const* W, const u16* wq, const u16* wop,
                         const u16* wf1p, const u16* wf2p, float* outp) {
    const long long M = NB * L;
    run_gemm<false, 1, true>(stream, inB, ND, 0, wq, 3 * ND, 0, qkvb, 3 * ND, 0,
                             W[1], nullptr, 0, 0, M, 3 * ND, (int)ND, 1, 1.0f);
    for (int b = 0; b < (int)NB; ++b) {
      const u16* qb = qkvb + (long long)b * L * 3 * ND;
      run_gemm<true, 0, false>(stream, qb, 3 * ND, NDH, qb + ND, 3 * ND, NDH,
                               score, L, L * L, nullptr, nullptr, 0, 0,
                               L, L, (int)NDH, (int)NH, ATT_SCALE);
      softmax_rows<<<dim3((unsigned)(NH * L)), dim3(256), 0, stream>>>(
          score, pbuf, mask + (long long)b * L, (int)L, (int)(NH * L), 0);
      run_gemm<false, 0, true>(stream, pbuf, L, L * L, qb + 2 * ND, 3 * ND, NDH,
                               obuf + (long long)b * L * ND, ND, NDH,
                               nullptr, nullptr, 0, 0, L, NDH, (int)L, (int)NH, 1.0f);
    }
    run_gemm<false, 2, false>(stream, obuf, ND, 0, wop, ND, 0, pre, ND, 0,
                              W[3], inF, ND, 0, M, ND, (int)ND, 1, 1.0f);
    layernorm_rows<<<dim3((unsigned)M), dim3(256), 0, stream>>>(pre, W[4], W[5], lnf, lnb, (int)ND);
    run_gemm<false, 3, true>(stream, lnb, ND, 0, wf1p, ND, 0, fbuf, ND, 0,
                             W[7], nullptr, 0, 0, M, ND, (int)ND, 1, 1.0f);
    run_gemm<false, 2, false>(stream, fbuf, ND, 0, wf2p, ND, 0, pre, ND, 0,
                              W[9], lnf, ND, 0, M, ND, (int)ND, 1, 1.0f);
    layernorm_rows<<<dim3((unsigned)M), dim3(256), 0, stream>>>(pre, W[10], W[11], outp, nullptr, (int)ND);
  };

  float* outx = (float*)d_out;
  float* outy = outx + NB * NLX * ND;
  run_encoder(xo1, xo1b, xm, NLX, P[0], wqkv[0], wo[0], wf1[0], wf2[0], outx);
  run_encoder(yo1, yo1b, ym, NLY, P[1], wqkv[1], wo[1], wf1[1], wf2[1], outy);
}
